// TransformerClsDecoderLayer_45011257262214
// MI455X (gfx1250) — matrix operand order repaired and checked by reading, not compiled, not run on hardware
//
#include <hip/hip_runtime.h>
#include <hip/hip_bf16.h>

// ---------------------------------------------------------------------------
// TransformerClsDecoderLayer for MI455X (gfx1250, wave32, WMMA).
// All GEMMs run as bf16 x bf16 -> f32 via v_wmma_f32_16x16x32_bf16.
// Softmax / LayerNorm / residuals in fp32.
// Workspace requirement: ~765 MB (logits f32 + probs bf16 dominate).
// ---------------------------------------------------------------------------

#define NQ_  300
#define BS_  16
#define HW_  2048
#define DM_  256
#define NH_  8
#define DFF_ 2048

typedef unsigned short u16;
typedef __attribute__((ext_vector_type(8)))  __bf16 v8bf;
typedef __attribute__((ext_vector_type(16))) __bf16 v16bf;
typedef __attribute__((ext_vector_type(8)))  float  v8f;

union FragBF { v16bf v; v8bf h[2]; };

__device__ __forceinline__ u16 f2bf(float f) {
  unsigned int u = __float_as_uint(f);
  u += 0x7FFFu + ((u >> 16) & 1u);            // round-to-nearest-even
  return (u16)(u >> 16);
}
__device__ __forceinline__ float bf2f(u16 h) {
  return __uint_as_float(((unsigned int)h) << 16);
}

__device__ __forceinline__ void storeOut(float* C, long idx, float v) { C[idx] = v; }
__device__ __forceinline__ void storeOut(u16*   C, long idx, float v) { C[idx] = f2bf(v); }

// ---------------------------------------------------------------------------
// Batched NT GEMM: C[b,h] = act( (A[b,h] * B[b,h]^T + bias) * scale )
//   A: M x K row-major (lda elements/row), B: N x K row-major (ldb), bf16.
//   Per-z offsets: z -> (zb = z / Hd, zh = z % Hd), off = zb*xB + zh*xH.
//   Block = 128 threads (4 waves), 64x64 tile; each wave computes 32x32 via
//   four 16x16x32 bf16 WMMAs per K step.
// A-fragment per ISA 16-bit 16x32 layout: lane<16 holds K=kadd+{0..7,16..23};
// B-fragment: same K order as A (lane<16 holds K=kadd+{0..7,16..23}).
// ---------------------------------------------------------------------------
template <typename OutT>
__global__ void __launch_bounds__(128)
gemm_bf16_nt(const u16* __restrict__ A, long lda, long aBb, long aHh,
             const u16* __restrict__ B, long ldb, long bBb, long bHh,
             const float* __restrict__ bias,
             OutT* __restrict__ C, long ldc, long cBb, long cHh,
             int M, int N, int K, float scale, int relu, int Hd)
{
  const int z  = blockIdx.z;
  const int zb = z / Hd;
  const int zh = z % Hd;
  const u16* Ab = A + (long)zb * aBb + (long)zh * aHh;
  const u16* Bb = B + (long)zb * bBb + (long)zh * bHh;
  const long cOff = (long)zb * cBb + (long)zh * cHh;

  const int lane = threadIdx.x & 31;
  const int wave = threadIdx.x >> 5;
  const int tileM = blockIdx.y * 64 + ((wave >> 1) << 5);
  const int tileN = blockIdx.x * 64 + ((wave & 1) << 5);

  const int l15 = lane & 15;
  const int hi  = lane >> 4;          // 0 or 1 (lane half)

  // Fragment source rows (clamped at edges; clamped lanes produce garbage
  // that is never stored because stores are bounds-guarded).
  int ra0 = tileM + l15;      ra0 = ra0 < M ? ra0 : M - 1;
  int ra1 = tileM + 16 + l15; ra1 = ra1 < M ? ra1 : M - 1;
  int rb0 = tileN + l15;      rb0 = rb0 < N ? rb0 : N - 1;
  int rb1 = tileN + 16 + l15; rb1 = rb1 < N ? rb1 : N - 1;

  const u16* pa0 = Ab + (long)ra0 * lda + (hi << 3);   // A: +0/+8, then +16
  const u16* pa1 = Ab + (long)ra1 * lda + (hi << 3);
  const u16* pb0 = Bb + (long)rb0 * ldb + (hi << 3);   // B: +0/+8, then +16
  const u16* pb1 = Bb + (long)rb1 * ldb + (hi << 3);

  const v8f z8 = {0.f, 0.f, 0.f, 0.f, 0.f, 0.f, 0.f, 0.f};
  v8f acc00 = z8, acc01 = z8, acc10 = z8, acc11 = z8;

  for (int k0 = 0; k0 < K; k0 += 32) {
    FragBF a0, a1, b0, b1;
    a0.h[0] = *(const v8bf*)(pa0 + k0);
    a0.h[1] = *(const v8bf*)(pa0 + k0 + 16);
    a1.h[0] = *(const v8bf*)(pa1 + k0);
    a1.h[1] = *(const v8bf*)(pa1 + k0 + 16);
    b0.h[0] = *(const v8bf*)(pb0 + k0);
    b0.h[1] = *(const v8bf*)(pb0 + k0 + 16);
    b1.h[0] = *(const v8bf*)(pb1 + k0);
    b1.h[1] = *(const v8bf*)(pb1 + k0 + 16);
    acc00 = __builtin_amdgcn_wmma_f32_16x16x32_bf16(false, a0.v, false, b0.v,
                                                    (short)0, acc00, false, false);
    acc01 = __builtin_amdgcn_wmma_f32_16x16x32_bf16(false, a0.v, false, b1.v,
                                                    (short)0, acc01, false, false);
    acc10 = __builtin_amdgcn_wmma_f32_16x16x32_bf16(false, a1.v, false, b0.v,
                                                    (short)0, acc10, false, false);
    acc11 = __builtin_amdgcn_wmma_f32_16x16x32_bf16(false, a1.v, false, b1.v,
                                                    (short)0, acc11, false, false);
  }

  v8f accs[2][2] = {{acc00, acc01}, {acc10, acc11}};
  #pragma unroll
  for (int i = 0; i < 2; ++i) {
    #pragma unroll
    for (int j = 0; j < 2; ++j) {
      const int n = tileN + j * 16 + l15;
      if (n >= N) continue;
      const float bvv = bias ? bias[n] : 0.0f;
      const int mBase = tileM + i * 16 + (hi << 3);
      #pragma unroll
      for (int r = 0; r < 8; ++r) {
        const int m = mBase + r;
        if (m >= M) continue;
        float vv = (accs[i][j][r] + bvv) * scale;
        if (relu) vv = vv > 0.f ? vv : 0.f;
        storeOut(C, cOff + (long)m * ldc + n, vv);
      }
    }
  }
}

// --------------------------- elementwise kernels ---------------------------

__global__ void pack_bf_kernel(const float* __restrict__ x, u16* __restrict__ y, long n) {
  long i = (long)blockIdx.x * blockDim.x + threadIdx.x;
  if (i < n) y[i] = f2bf(x[i]);
}

// w: [K, N] f32 -> wT: [N, K] bf16
__global__ void transpose_pack_kernel(const float* __restrict__ w, u16* __restrict__ wT,
                                      int K, int N) {
  long i = (long)blockIdx.x * blockDim.x + threadIdx.x;
  long total = (long)K * N;
  if (i >= total) return;
  long n = i / K;
  long k = i % K;
  wT[i] = f2bf(w[k * (long)N + n]);
}

// q2[(q*BS+b), h*64 + d] = d<32 ? qc+qp (at h*32+d) : qs (at h*32+d-32)
__global__ void assemble_q_kernel(const float* __restrict__ qc, const float* __restrict__ qp,
                                  const float* __restrict__ qs, u16* __restrict__ q2,
                                  long rows) {
  long i = (long)blockIdx.x * blockDim.x + threadIdx.x;
  if (i >= rows * 512) return;
  long row = i >> 9;
  int col = (int)(i & 511);
  int h = col >> 6, d = col & 63;
  float v;
  if (d < 32) v = qc[row * DM_ + h * 32 + d] + qp[row * DM_ + h * 32 + d];
  else        v = qs[row * DM_ + h * 32 + (d - 32)];
  q2[i] = f2bf(v);
}

__global__ void assemble_k_kernel(const float* __restrict__ kc, const float* __restrict__ kp,
                                  u16* __restrict__ k2, long rows) {
  long i = (long)blockIdx.x * blockDim.x + threadIdx.x;
  if (i >= rows * 512) return;
  long row = i >> 9;
  int col = (int)(i & 511);
  int h = col >> 6, d = col & 63;
  float v;
  if (d < 32) v = kc[row * DM_ + h * 32 + d] + kp[row * DM_ + h * 32 + d];
  else        v = kp[row * DM_ + h * 32 + (d - 32)];
  k2[i] = f2bf(v);
}

// vhT[b, n, k] = vh[(k*BS+b), n]   (for NT ctx GEMM: B rows along n, K contiguous)
__global__ void transpose_vh_kernel(const u16* __restrict__ vh, u16* __restrict__ vhT) {
  long i = (long)blockIdx.x * blockDim.x + threadIdx.x;
  long total = (long)BS_ * DM_ * HW_;
  if (i >= total) return;
  int k = (int)(i % HW_);
  long t = i / HW_;
  int n = (int)(t % DM_);
  int b = (int)(t / DM_);
  vhT[i] = vh[((long)k * BS_ + b) * DM_ + n];
}

// One block per (b, q): softmax over HW for all 8 heads; writes bf16 probs and
// the head-averaged attention (second tuple output) directly.
__global__ void __launch_bounds__(256)
softmax_attn_kernel(const float* __restrict__ logits, u16* __restrict__ probs,
                    float* __restrict__ attn_avg) {
  const int b = blockIdx.x / NQ_;
  const int q = blockIdx.x % NQ_;
  const int tid = threadIdx.x;
  __shared__ float red[256];
  float avg[8];
  #pragma unroll
  for (int r = 0; r < 8; ++r) avg[r] = 0.f;

  for (int h = 0; h < NH_; ++h) {
    const long off = (((long)(b * NH_ + h)) * NQ_ + q) * HW_;
    const float* row = logits + off;
    float loc[8];
    float mx = -3.4e38f;
    #pragma unroll
    for (int r = 0; r < 8; ++r) { loc[r] = row[r * 256 + tid]; mx = fmaxf(mx, loc[r]); }
    red[tid] = mx; __syncthreads();
    for (int s = 128; s > 0; s >>= 1) {
      if (tid < s) red[tid] = fmaxf(red[tid], red[tid + s]);
      __syncthreads();
    }
    mx = red[0]; __syncthreads();
    float sum = 0.f;
    #pragma unroll
    for (int r = 0; r < 8; ++r) { loc[r] = __expf(loc[r] - mx); sum += loc[r]; }
    red[tid] = sum; __syncthreads();
    for (int s = 128; s > 0; s >>= 1) {
      if (tid < s) red[tid] += red[tid + s];
      __syncthreads();
    }
    const float inv = 1.f / red[0]; __syncthreads();
    u16* prow = probs + off;
    #pragma unroll
    for (int r = 0; r < 8; ++r) {
      float p = loc[r] * inv;
      prow[r * 256 + tid] = f2bf(p);
      avg[r] += p * (1.0f / NH_);
    }
  }
  float* arow = attn_avg + ((long)(b * NQ_ + q)) * HW_;
  #pragma unroll
  for (int r = 0; r < 8; ++r) arow[r * 256 + tid] = avg[r];
}

// y = LayerNorm(a + c) * g + be ; optional bf16 copy for next GEMM.
// One block (256 threads) per row; D == 256.
__global__ void __launch_bounds__(256)
add_ln_kernel(const float* __restrict__ a, const float* __restrict__ c,
              const float* __restrict__ g, const float* __restrict__ be,
              float* __restrict__ y, u16* __restrict__ ybf) {
  const long row = blockIdx.x;
  const int tid = threadIdx.x;
  const long i = row * DM_ + tid;
  const float x = a[i] + c[i];
  __shared__ float red[256];
  red[tid] = x; __syncthreads();
  for (int s = 128; s > 0; s >>= 1) {
    if (tid < s) red[tid] += red[tid + s];
    __syncthreads();
  }
  const float mean = red[0] * (1.f / DM_); __syncthreads();
  const float d = x - mean;
  red[tid] = d * d; __syncthreads();
  for (int s = 128; s > 0; s >>= 1) {
    if (tid < s) red[tid] += red[tid + s];
    __syncthreads();
  }
  const float var = red[0] * (1.f / DM_);
  const float o = d * rsqrtf(var + 1e-5f) * g[tid] + be[tid];
  y[i] = o;
  if (ybf) ybf[i] = f2bf(o);
}

// ---------------------------------------------------------------------------

extern "C" void kernel_launch(void* const* d_in, const int* in_sizes, int n_in,
                              void* d_out, int out_size, void* d_ws, size_t ws_size,
                              hipStream_t stream) {
  (void)in_sizes; (void)n_in; (void)out_size; (void)ws_size;

  const float* tgt  = (const float*)d_in[0];
  const float* mem  = (const float*)d_in[1];
  const float* pos  = (const float*)d_in[2];
  const float* qpos = (const float*)d_in[3];
  const float* sine = (const float*)d_in[4];
  const float* w_qc = (const float*)d_in[5];  const float* b_qc = (const float*)d_in[6];
  const float* w_qp = (const float*)d_in[7];  const float* b_qp = (const float*)d_in[8];
  const float* w_kc = (const float*)d_in[9];  const float* b_kc = (const float*)d_in[10];
  const float* w_kp = (const float*)d_in[11]; const float* b_kp = (const float*)d_in[12];
  const float* w_v  = (const float*)d_in[13]; const float* b_v  = (const float*)d_in[14];
  const float* w_qs = (const float*)d_in[15]; const float* b_qs = (const float*)d_in[16];
  const float* wq   = (const float*)d_in[17]; const float* bq   = (const float*)d_in[18];
  const float* wk   = (const float*)d_in[19]; const float* bk   = (const float*)d_in[20];
  const float* wv   = (const float*)d_in[21]; const float* bv   = (const float*)d_in[22];
  const float* wo   = (const float*)d_in[23]; const float* bo   = (const float*)d_in[24];
  const float* w1   = (const float*)d_in[25]; const float* b1   = (const float*)d_in[26];
  const float* w2   = (const float*)d_in[27]; const float* b2   = (const float*)d_in[28];
  const float* g2   = (const float*)d_in[29]; const float* be2  = (const float*)d_in[30];
  const float* g3   = (const float*)d_in[31]; const float* be3  = (const float*)d_in[32];

  float* out_x    = (float*)d_out;                       // [NQ, BS, D] f32
  float* out_attn = out_x + (long)NQ_ * BS_ * DM_;       // [BS, NQ, HW] f32

  // --- bump allocator over d_ws ---
  char* base = (char*)d_ws;
  size_t off = 0;
  auto alloc = [&](size_t bytes) -> void* {
    void* p = base + off;
    off = (off + bytes + 255) & ~(size_t)255;
    return p;
  };

  const long NQB = (long)NQ_ * BS_;   // 4800
  const long HWB = (long)HW_ * BS_;   // 32768
  const int  D2  = 2 * DM_;           // 512

  // bf16 activation inputs
  u16* tgt_bf  = (u16*)alloc(NQB * DM_ * 2);
  u16* mem_bf  = (u16*)alloc(HWB * DM_ * 2);
  u16* pos_bf  = (u16*)alloc(HWB * DM_ * 2);
  u16* qpos_bf = (u16*)alloc(NQB * DM_ * 2);
  u16* sine_bf = (u16*)alloc(NQB * DM_ * 2);
  // transposed bf16 weights [N, K]
  u16* wqcT = (u16*)alloc((long)DM_ * DM_ * 2);
  u16* wqpT = (u16*)alloc((long)DM_ * DM_ * 2);
  u16* wkcT = (u16*)alloc((long)DM_ * DM_ * 2);
  u16* wkpT = (u16*)alloc((long)DM_ * DM_ * 2);
  u16* wvinT= (u16*)alloc((long)DM_ * DM_ * 2);   // w_v
  u16* wqsT = (u16*)alloc((long)DM_ * DM_ * 2);
  u16* wqT  = (u16*)alloc((long)D2 * D2 * 2);
  u16* wkT  = (u16*)alloc((long)D2 * D2 * 2);
  u16* wvT  = (u16*)alloc((long)DM_ * DM_ * 2);   // wv (mha value proj)
  u16* woT  = (u16*)alloc((long)DM_ * DM_ * 2);
  u16* w1T  = (u16*)alloc((long)DFF_ * DM_ * 2);
  u16* w2T  = (u16*)alloc((long)DM_ * DFF_ * 2);
  // f32 intermediates
  float* qc_f = (float*)alloc(NQB * DM_ * 4);
  float* qp_f = (float*)alloc(NQB * DM_ * 4);
  float* qs_f = (float*)alloc(NQB * DM_ * 4);
  float* kc_f = (float*)alloc(HWB * DM_ * 4);
  float* kp_f = (float*)alloc(HWB * DM_ * 4);
  u16*   vin_bf = (u16*)alloc(HWB * DM_ * 2);
  u16*   q2_bf  = (u16*)alloc(NQB * D2 * 2);
  u16*   k2_bf  = (u16*)alloc(HWB * D2 * 2);
  u16*   qh_bf  = (u16*)alloc(NQB * D2 * 2);
  u16*   kh_bf  = (u16*)alloc(HWB * D2 * 2);
  u16*   vh_bf  = (u16*)alloc(HWB * DM_ * 2);
  u16*   vhT_bf = (u16*)alloc((long)BS_ * DM_ * HW_ * 2);
  float* logits_f = (float*)alloc((long)BS_ * NH_ * NQ_ * HW_ * 4);  // 315 MB
  u16*   probs_bf = (u16*)alloc((long)BS_ * NH_ * NQ_ * HW_ * 2);    // 157 MB
  u16*   ctx_bf  = (u16*)alloc(NQB * DM_ * 2);
  float* tgt2_f  = (float*)alloc(NQB * DM_ * 4);
  float* x_f     = (float*)alloc(NQB * DM_ * 4);
  u16*   x_bf    = (u16*)alloc(NQB * DM_ * 2);
  u16*   ffn1_bf = (u16*)alloc(NQB * DFF_ * 2);
  float* ffn2_f  = (float*)alloc(NQB * DM_ * 4);

  auto pack = [&](const float* x, u16* y, long n) {
    pack_bf_kernel<<<dim3((unsigned)((n + 255) / 256)), dim3(256), 0, stream>>>(x, y, n);
  };
  auto tpack = [&](const float* w, u16* wT, int K, int N) {
    long t = (long)K * N;
    transpose_pack_kernel<<<dim3((unsigned)((t + 255) / 256)), dim3(256), 0, stream>>>(w, wT, K, N);
  };
  auto gemmF = [&](const u16* A, long lda, long aB, long aH,
                   const u16* B, long ldb, long bB, long bH,
                   const float* bias, float* C, long ldc, long cB, long cH,
                   int M, int N, int K, float scale, int relu, int Hd, int batches) {
    dim3 g((unsigned)((N + 63) / 64), (unsigned)((M + 63) / 64), (unsigned)batches);
    gemm_bf16_nt<float><<<g, dim3(128), 0, stream>>>(A, lda, aB, aH, B, ldb, bB, bH,
                                                     bias, C, ldc, cB, cH, M, N, K,
                                                     scale, relu, Hd);
  };
  auto gemmB = [&](const u16* A, long lda, long aB, long aH,
                   const u16* B, long ldb, long bB, long bH,
                   const float* bias, u16* C, long ldc, long cB, long cH,
                   int M, int N, int K, float scale, int relu, int Hd, int batches) {
    dim3 g((unsigned)((N + 63) / 64), (unsigned)((M + 63) / 64), (unsigned)batches);
    gemm_bf16_nt<u16><<<g, dim3(128), 0, stream>>>(A, lda, aB, aH, B, ldb, bB, bH,
                                                   bias, C, ldc, cB, cH, M, N, K,
                                                   scale, relu, Hd);
  };

  // 1. convert activations + weights to bf16 (weights transposed to [N,K])
  pack(tgt,  tgt_bf,  NQB * DM_);
  pack(mem,  mem_bf,  HWB * DM_);
  pack(pos,  pos_bf,  HWB * DM_);
  pack(qpos, qpos_bf, NQB * DM_);
  pack(sine, sine_bf, NQB * DM_);
  tpack(w_qc, wqcT, DM_, DM_);  tpack(w_qp, wqpT, DM_, DM_);
  tpack(w_kc, wkcT, DM_, DM_);  tpack(w_kp, wkpT, DM_, DM_);
  tpack(w_v,  wvinT, DM_, DM_); tpack(w_qs, wqsT, DM_, DM_);
  tpack(wq, wqT, D2, D2);       tpack(wk, wkT, D2, D2);
  tpack(wv, wvT, DM_, DM_);     tpack(wo, woT, DM_, DM_);
  tpack(w1, w1T, DM_, DFF_);    tpack(w2, w2T, DFF_, DM_);

  // 2. conditional cross-attn projections
  gemmF(tgt_bf,  DM_,0,0, wqcT, DM_,0,0, b_qc, qc_f, DM_,0,0, (int)NQB, DM_, DM_, 1.f,0, 1, 1);
  gemmF(qpos_bf, DM_,0,0, wqpT, DM_,0,0, b_qp, qp_f, DM_,0,0, (int)NQB, DM_, DM_, 1.f,0, 1, 1);
  gemmF(sine_bf, DM_,0,0, wqsT, DM_,0,0, b_qs, qs_f, DM_,0,0, (int)NQB, DM_, DM_, 1.f,0, 1, 1);
  gemmF(mem_bf,  DM_,0,0, wkcT, DM_,0,0, b_kc, kc_f, DM_,0,0, (int)HWB, DM_, DM_, 1.f,0, 1, 1);
  gemmF(pos_bf,  DM_,0,0, wkpT, DM_,0,0, b_kp, kp_f, DM_,0,0, (int)HWB, DM_, DM_, 1.f,0, 1, 1);
  gemmB(mem_bf,  DM_,0,0, wvinT,DM_,0,0, b_v,  vin_bf, DM_,0,0, (int)HWB, DM_, DM_, 1.f,0, 1, 1);

  // 3. assemble concatenated q / k (head-interleaved [.., H, 64])
  assemble_q_kernel<<<dim3((unsigned)((NQB * 512 + 255) / 256)), dim3(256), 0, stream>>>(
      qc_f, qp_f, qs_f, q2_bf, NQB);
  assemble_k_kernel<<<dim3((unsigned)((HWB * 512 + 255) / 256)), dim3(256), 0, stream>>>(
      kc_f, kp_f, k2_bf, HWB);

  // 4. MHA internal projections; scale folded into qh
  const float SCALE = 0.125f;  // 1/sqrt(64)
  gemmB(q2_bf, D2,0,0, wqT, D2,0,0, bq, qh_bf, D2,0,0, (int)NQB, D2, D2, SCALE,0, 1, 1);
  gemmB(k2_bf, D2,0,0, wkT, D2,0,0, bk, kh_bf, D2,0,0, (int)HWB, D2, D2, 1.f,  0, 1, 1);
  gemmB(vin_bf,DM_,0,0, wvT, DM_,0,0, bv, vh_bf, DM_,0,0, (int)HWB, DM_, DM_, 1.f,0, 1, 1);
  transpose_vh_kernel<<<dim3((unsigned)(((long)BS_*DM_*HW_ + 255) / 256)), dim3(256), 0, stream>>>(
      vh_bf, vhT_bf);

  // 5. logits[b,h,q,k] = sum_d qh * kh   (128 batched NT GEMMs)
  gemmF(qh_bf, (long)BS_ * D2, D2, 64,
        kh_bf, (long)BS_ * D2, D2, 64,
        nullptr, logits_f, HW_, (long)NH_ * NQ_ * HW_, (long)NQ_ * HW_,
        NQ_, HW_, 64, 1.f, 0, NH_, BS_ * NH_);

  // 6. softmax + head-average (writes attn_avg output)
  softmax_attn_kernel<<<dim3((unsigned)(BS_ * NQ_)), dim3(256), 0, stream>>>(
      logits_f, probs_bf, out_attn);

  // 7. ctx[q,b,h*32+d] = sum_k probs * vh   (128 batched NT GEMMs vs vhT)
  gemmB(probs_bf, HW_, (long)NH_ * NQ_ * HW_, (long)NQ_ * HW_,
        vhT_bf,   HW_, (long)DM_ * HW_,       (long)32 * HW_,
        nullptr, ctx_bf, (long)BS_ * DM_, DM_, 32,
        NQ_, 32, HW_, 1.f, 0, NH_, BS_ * NH_);

  // 8. output projection, residual + LN1
  gemmF(ctx_bf, DM_,0,0, woT, DM_,0,0, bo, tgt2_f, DM_,0,0, (int)NQB, DM_, DM_, 1.f,0, 1, 1);
  add_ln_kernel<<<dim3((unsigned)NQB), dim3(256), 0, stream>>>(tgt, tgt2_f, g2, be2, x_f, x_bf);

  // 9. FFN + residual + LN2 (final x output)
  gemmB(x_bf,   DM_,0,0, w1T, DM_,0,0, b1, ffn1_bf, DFF_,0,0, (int)NQB, DFF_, DM_, 1.f,1, 1, 1);
  gemmF(ffn1_bf,DFF_,0,0, w2T, DFF_,0,0, b2, ffn2_f, DM_,0,0, (int)NQB, DM_, DFF_, 1.f,0, 1, 1);
  add_ln_kernel<<<dim3((unsigned)NQB), dim3(256), 0, stream>>>(x_f, ffn2_f, g3, be3, out_x, nullptr);
}
